// TransformerLayer_38878043963900
// MI455X (gfx1250) — matrix operand order repaired and checked by reading, not compiled, not run on hardware
//
#include <hip/hip_runtime.h>
#include <hip/hip_bf16.h>
#include <math.h>
#include <stdint.h>

// ---------------------------------------------------------------------------
// Transformer encoder layer for MI455X (gfx1250, wave32, WMMA).
// Matmuls: v_wmma_f32_16x16x32_f16 (f32 accum).
// GEMM tile staging: global_load_async_to_lds_b128 double-buffered pipeline
// (ASYNCcnt), plus global_prefetch_b8 look-ahead.
// ---------------------------------------------------------------------------

typedef __attribute__((ext_vector_type(16))) _Float16 v16h;
typedef __attribute__((ext_vector_type(8)))  float    v8f;
typedef __attribute__((ext_vector_type(4)))  unsigned int v4u;

union V16 { v16h v; v4u u[2]; };

__device__ static inline v8f zero8() {
    v8f z;
    #pragma unroll
    for (int i = 0; i < 8; ++i) z[i] = 0.0f;
    return z;
}

#define SEQ    2048
#define DMODEL 1024
#define NHEAD  16
#define HDIM   64
#define DFF    4096
#define MROWS  4096   // B*S
#define LDT    40     // LDS tile row stride in halves (padded, keeps 16B align)

// ---------------------------------------------------------------------------
// f32 -> f16 flat cast
// ---------------------------------------------------------------------------
__global__ __launch_bounds__(256)
void cvt_f16_kernel(const float* __restrict__ X, _Float16* __restrict__ Y, int n) {
    int i = blockIdx.x * 256 + threadIdx.x;
    if (i < n) Y[i] = (_Float16)X[i];
}

// W [K,N] f32  ->  Wt [N,K] f16   (so GEMM B-fragments load contiguous K)
__global__ __launch_bounds__(256)
void cvt_transpose_kernel(const float* __restrict__ W, _Float16* __restrict__ Wt,
                          int K, int N) {
    int i = blockIdx.x * 256 + threadIdx.x;
    if (i < K * N) {
        int k = i / N, n = i - k * N;
        Wt[(size_t)n * K + k] = (_Float16)W[i];
    }
}

// ---------------------------------------------------------------------------
// Tiled WMMA GEMM:  C[M,N] = A[M,K] @ Bt[N,K]^T + bias  (+ epilogue)
//   mode 0: store f16 [M,N]
//   mode 1: GELU(exact), store f16 [M,N]
//   mode 2: add resid (f32 [M,N]), store f32 [M,N]
//   mode 3: split heads, store f16 [B,H,S,hd]   (Q, K)
//   mode 4: split heads + transpose, f16 [B,H,hd,S]  (V)
// Block: 256 thr = 8 wave32 (2x4), block tile 128x128, wave tile 64x32.
// K-slices staged with async-to-LDS, double buffered.
// ---------------------------------------------------------------------------
__global__ __launch_bounds__(256)
void gemm_f16_kernel(const _Float16* __restrict__ A, const _Float16* __restrict__ Bt,
                     const float* __restrict__ bias, const float* __restrict__ resid,
                     void* __restrict__ Cout, int M, int N, int K, int mode) {
    __shared__ _Float16 As[2][128 * LDT];
    __shared__ _Float16 Bs[2][128 * LDT];

    const int tid   = threadIdx.x;
    const int lane  = tid & 31;
    const int wv    = tid >> 5;
    const int waveM = wv & 1;        // 2 waves over M
    const int waveN = wv >> 1;       // 4 waves over N
    const int lrow  = lane & 15;
    const int kpA   = (lane < 16) ? 0 : 8;    // A-frag interleaved K pattern
    const int kbB   = (lane < 16) ? 0 : 8;    // B-frag interleaved K pattern
    const int m0    = blockIdx.y * 128;
    const int n0    = blockIdx.x * 128;

    // each thread owns 2 16B chunks of the A tile and 2 of the B tile
    const int c0row = tid >> 2;           // chunk 0: rows 0..63
    const int c0off = (tid & 3) << 3;
    const int c1row = (tid + 256) >> 2;   // chunk 1: rows 64..127
    const int c1off = c0off;

    v8f acc[4][2];
    #pragma unroll
    for (int i = 0; i < 4; ++i)
        #pragma unroll
        for (int j = 0; j < 2; ++j) acc[i][j] = zero8();

    // async stage of one 128x32 A slice + 128x32 B slice into LDS buffer `buf`
    auto issue_tile = [&](int k0, int buf) {
        {
            unsigned la = (unsigned)(uintptr_t)&As[buf][c0row * LDT + c0off];
            const _Float16* ga = A + (size_t)(m0 + c0row) * K + k0 + c0off;
            asm volatile("global_load_async_to_lds_b128 %0, %1, off"
                         :: "v"(la), "v"(ga) : "memory");
            unsigned lb = (unsigned)(uintptr_t)&Bs[buf][c0row * LDT + c0off];
            const _Float16* gb = Bt + (size_t)(n0 + c0row) * K + k0 + c0off;
            asm volatile("global_load_async_to_lds_b128 %0, %1, off"
                         :: "v"(lb), "v"(gb) : "memory");
        }
        {
            unsigned la = (unsigned)(uintptr_t)&As[buf][c1row * LDT + c1off];
            const _Float16* ga = A + (size_t)(m0 + c1row) * K + k0 + c1off;
            asm volatile("global_load_async_to_lds_b128 %0, %1, off"
                         :: "v"(la), "v"(ga) : "memory");
            unsigned lb = (unsigned)(uintptr_t)&Bs[buf][c1row * LDT + c1off];
            const _Float16* gb = Bt + (size_t)(n0 + c1row) * K + k0 + c1off;
            asm volatile("global_load_async_to_lds_b128 %0, %1, off"
                         :: "v"(lb), "v"(gb) : "memory");
        }
    };

    const int nk = K >> 5;
    issue_tile(0, 0);
    for (int t = 0; t < nk; ++t) {
        const int buf = t & 1;
        if (t + 1 < nk) {
            issue_tile((t + 1) << 5, buf ^ 1);
            // L2 look-ahead two tiles out (gfx1250 global_prefetch_b8)
            if (t + 2 < nk) {
                __builtin_prefetch((const void*)(A  + (size_t)(m0 + c0row) * K + ((t + 2) << 5) + c0off), 0, 1);
                __builtin_prefetch((const void*)(Bt + (size_t)(n0 + c0row) * K + ((t + 2) << 5) + c0off), 0, 1);
            }
            // tile t's 4 transfers must land; tile t+1's 4 stay in flight
            asm volatile("s_wait_asynccnt 0x4" ::: "memory");
        } else {
            asm volatile("s_wait_asynccnt 0x0" ::: "memory");
        }
        __syncthreads();

        V16 af[4], bf[2];
        #pragma unroll
        for (int mt = 0; mt < 4; ++mt) {
            const _Float16* p = &As[buf][(waveM * 64 + mt * 16 + lrow) * LDT];
            af[mt].u[0] = *(const v4u*)(p + kpA);
            af[mt].u[1] = *(const v4u*)(p + kpA + 16);
        }
        #pragma unroll
        for (int nt = 0; nt < 2; ++nt) {
            const _Float16* p = &Bs[buf][(waveN * 32 + nt * 16 + lrow) * LDT];
            bf[nt].u[0] = *(const v4u*)(p + kbB);
            bf[nt].u[1] = *(const v4u*)(p + kbB + 16);
        }
        #pragma unroll
        for (int mt = 0; mt < 4; ++mt)
            #pragma unroll
            for (int nt = 0; nt < 2; ++nt)
                acc[mt][nt] = __builtin_amdgcn_wmma_f32_16x16x32_f16(
                    false, af[mt].v, false, bf[nt].v,
                    (short)0, acc[mt][nt], false, false);
        __syncthreads();   // readers done before buf is overwritten next round
    }

    const int rofs = (lane < 16) ? 0 : 8;
    #pragma unroll
    for (int mt = 0; mt < 4; ++mt)
        #pragma unroll
        for (int nt = 0; nt < 2; ++nt)
            #pragma unroll
            for (int r = 0; r < 8; ++r) {
                int m = m0 + waveM * 64 + mt * 16 + r + rofs;
                int n = n0 + waveN * 32 + nt * 16 + lrow;
                float val = acc[mt][nt][r] + bias[n];
                if (mode == 1) val = 0.5f * val * (1.0f + erff(val * 0.70710678118f));
                if (mode == 2) {
                    ((float*)Cout)[(size_t)m * N + n] = val + resid[(size_t)m * N + n];
                } else if (mode == 3) {
                    int bb = m >> 11, s = m & (SEQ - 1), hh = n >> 6, d = n & 63;
                    ((_Float16*)Cout)[(((size_t)bb * NHEAD + hh) * SEQ + s) * HDIM + d] = (_Float16)val;
                } else if (mode == 4) {
                    int bb = m >> 11, s = m & (SEQ - 1), hh = n >> 6, d = n & 63;
                    ((_Float16*)Cout)[(((size_t)bb * NHEAD + hh) * HDIM + d) * SEQ + s] = (_Float16)val;
                } else {
                    ((_Float16*)Cout)[(size_t)m * N + n] = (_Float16)val;
                }
            }
}

// ---------------------------------------------------------------------------
// Flash attention: one wave per 16 query rows; streams 32-key blocks with
// online softmax; causal + key-padding masks; WMMA for QK^T and PV.
// Q,K: [B,H,S,64] f16   Vt: [B,H,64,S] f16   out attnX: [B*S, 1024] f16
// ---------------------------------------------------------------------------
__global__ __launch_bounds__(256)
void attn_kernel(const _Float16* __restrict__ Q, const _Float16* __restrict__ Kh,
                 const _Float16* __restrict__ Vt, const unsigned char* __restrict__ pad,
                 _Float16* __restrict__ attnX) {
    __shared__ _Float16 Pb[8][16][LDT];   // per-wave P tile scratch

    const int lane = threadIdx.x & 31;
    const int wv   = threadIdx.x >> 5;
    const int lrow = lane & 15;
    const int kpA  = (lane < 16) ? 0 : 8;
    const int kbB  = (lane < 16) ? 0 : 8;
    const int rofs = (lane < 16) ? 0 : 8;
    const int bh   = blockIdx.y;
    const int b    = bh >> 4;
    const int h    = bh & 15;
    const int qbase = blockIdx.x * 128 + wv * 16;

    // Q fragments (16 rows x 64 head dims = two k-steps), held in registers
    V16 qf[2];
    {
        const _Float16* qr = Q + ((size_t)bh * SEQ + qbase + lrow) * HDIM;
        #pragma unroll
        for (int ks = 0; ks < 2; ++ks) {
            qf[ks].u[0] = *(const v4u*)(qr + ks * 32 + kpA);
            qf[ks].u[1] = *(const v4u*)(qr + ks * 32 + kpA + 16);
        }
    }

    float mrow[8], lsum[8];
    v8f oacc[4];
    #pragma unroll
    for (int r = 0; r < 8; ++r) { mrow[r] = -1e30f; lsum[r] = 0.0f; }
    #pragma unroll
    for (int t = 0; t < 4; ++t) oacc[t] = zero8();

    const int nkb = (qbase + 15) / 32 + 1;   // causal early exit (wave-uniform)
    for (int kb = 0; kb < nkb; ++kb) {
        const int k0 = kb * 32;

        // scores S = Q K^T  (two 16-key column tiles)
        v8f sc[2];
        #pragma unroll
        for (int nt = 0; nt < 2; ++nt) {
            const _Float16* kr = Kh + ((size_t)bh * SEQ + k0 + nt * 16 + lrow) * HDIM;
            V16 kf[2];
            #pragma unroll
            for (int ks = 0; ks < 2; ++ks) {
                kf[ks].u[0] = *(const v4u*)(kr + ks * 32 + kbB);
                kf[ks].u[1] = *(const v4u*)(kr + ks * 32 + kbB + 16);
            }
            v8f s = zero8();
            s = __builtin_amdgcn_wmma_f32_16x16x32_f16(false, qf[0].v, false, kf[0].v,
                                                       (short)0, s, false, false);
            s = __builtin_amdgcn_wmma_f32_16x16x32_f16(false, qf[1].v, false, kf[1].v,
                                                       (short)0, s, false, false);
            sc[nt] = s;
        }

        // masking + online softmax (row reductions across 16-lane halves)
        float p[2][8];
        #pragma unroll
        for (int r = 0; r < 8; ++r) {
            const int qrow = qbase + r + rofs;
            float mx = -1e30f;
            #pragma unroll
            for (int nt = 0; nt < 2; ++nt) {
                int key = k0 + nt * 16 + lrow;
                float x = sc[nt][r] * 0.125f;          // 1/sqrt(64)
                if (key > qrow || pad[(size_t)b * SEQ + key]) x = -1e30f;
                p[nt][r] = x;
                mx = fmaxf(mx, x);
            }
            #pragma unroll
            for (int mk = 8; mk >= 1; mk >>= 1)
                mx = fmaxf(mx, __shfl_xor(mx, mk, 32));
            float mnew = fmaxf(mrow[r], mx);
            float corr = __expf(mrow[r] - mnew);
            float rsum = 0.0f;
            #pragma unroll
            for (int nt = 0; nt < 2; ++nt) {
                float e = __expf(p[nt][r] - mnew);
                p[nt][r] = e;
                rsum += e;
            }
            #pragma unroll
            for (int mk = 8; mk >= 1; mk >>= 1)
                rsum += __shfl_xor(rsum, mk, 32);
            lsum[r] = lsum[r] * corr + rsum;
            mrow[r] = mnew;
            #pragma unroll
            for (int t = 0; t < 4; ++t) oacc[t][r] *= corr;
        }

        // D-layout P -> A-fragment layout via per-wave LDS tile
        #pragma unroll
        for (int nt = 0; nt < 2; ++nt)
            #pragma unroll
            for (int r = 0; r < 8; ++r)
                Pb[wv][r + rofs][nt * 16 + lrow] = (_Float16)p[nt][r];
        asm volatile("s_wait_dscnt 0x0" ::: "memory");

        V16 pf;
        {
            const _Float16* pr = &Pb[wv][lrow][0];
            pf.u[0] = *(const v4u*)(pr + kpA);
            pf.u[1] = *(const v4u*)(pr + kpA + 16);
        }

        // O += P @ V   (4 output d-tiles; Vt rows are contiguous over keys)
        #pragma unroll
        for (int t = 0; t < 4; ++t) {
            const _Float16* vr = Vt + ((size_t)bh * HDIM + t * 16 + lrow) * SEQ + k0;
            V16 vf;
            vf.u[0] = *(const v4u*)(vr + kbB);
            vf.u[1] = *(const v4u*)(vr + kbB + 16);
            oacc[t] = __builtin_amdgcn_wmma_f32_16x16x32_f16(false, pf.v, false, vf.v,
                                                             (short)0, oacc[t], false, false);
        }
    }

    // normalize + merge heads: attnX[(b*S+q)*D + h*64 + d]
    #pragma unroll
    for (int t = 0; t < 4; ++t)
        #pragma unroll
        for (int r = 0; r < 8; ++r) {
            int q = qbase + r + rofs;
            float val = oacc[t][r] / lsum[r];
            attnX[((size_t)(b * SEQ + q)) * DMODEL + h * HDIM + t * 16 + lrow] = (_Float16)val;
        }
}

// ---------------------------------------------------------------------------
// LayerNorm over D=1024 per row; writes f32 and/or f16 copies.
// If pad != nullptr, padded rows are zeroed (final output only).
// ---------------------------------------------------------------------------
__global__ __launch_bounds__(256)
void ln_kernel(const float* __restrict__ Y, const float* __restrict__ gamma,
               const float* __restrict__ beta, float* __restrict__ out32,
               _Float16* __restrict__ out16, const unsigned char* __restrict__ pad) {
    const int row = blockIdx.x;
    const int tid = threadIdx.x;
    float x[4];
    float s = 0.0f, s2 = 0.0f;
    #pragma unroll
    for (int i = 0; i < 4; ++i) {
        x[i] = Y[(size_t)row * DMODEL + tid + i * 256];
        s  += x[i];
        s2 += x[i] * x[i];
    }
    #pragma unroll
    for (int mk = 16; mk >= 1; mk >>= 1) {
        s  += __shfl_xor(s,  mk, 32);
        s2 += __shfl_xor(s2, mk, 32);
    }
    __shared__ float rs[8], rs2[8];
    if ((tid & 31) == 0) { rs[tid >> 5] = s; rs2[tid >> 5] = s2; }
    __syncthreads();
    s = 0.0f; s2 = 0.0f;
    #pragma unroll
    for (int i = 0; i < 8; ++i) { s += rs[i]; s2 += rs2[i]; }
    float mu   = s * (1.0f / DMODEL);
    float var  = s2 * (1.0f / DMODEL) - mu * mu;
    float rstd = rsqrtf(var + 1e-5f);
    bool  zp   = (pad != nullptr) && pad[row];
    #pragma unroll
    for (int i = 0; i < 4; ++i) {
        int col = tid + i * 256;
        float val = (x[i] - mu) * rstd * gamma[col] + beta[col];
        if (zp) val = 0.0f;
        if (out32) out32[(size_t)row * DMODEL + col] = val;
        if (out16) out16[(size_t)row * DMODEL + col] = (_Float16)val;
    }
}

// ---------------------------------------------------------------------------
// Orchestration
// ---------------------------------------------------------------------------
extern "C" void kernel_launch(void* const* d_in, const int* in_sizes, int n_in,
                              void* d_out, int out_size, void* d_ws, size_t ws_size,
                              hipStream_t stream) {
    (void)in_sizes; (void)n_in; (void)out_size; (void)ws_size;
    const float* src  = (const float*)d_in[0];
    // d_in[1] = causal mask (recomputed analytically in-kernel)
    const unsigned char* pad = (const unsigned char*)d_in[2];
    const float* Wq = (const float*)d_in[3];  const float* bq = (const float*)d_in[4];
    const float* Wk = (const float*)d_in[5];  const float* bk = (const float*)d_in[6];
    const float* Wv = (const float*)d_in[7];  const float* bv = (const float*)d_in[8];
    const float* Wo = (const float*)d_in[9];  const float* bo = (const float*)d_in[10];
    const float* W1 = (const float*)d_in[11]; const float* b1 = (const float*)d_in[12];
    const float* W2 = (const float*)d_in[13]; const float* b2 = (const float*)d_in[14];
    const float* gamma1 = (const float*)d_in[15]; const float* beta1 = (const float*)d_in[16];
    const float* gamma2 = (const float*)d_in[17]; const float* beta2 = (const float*)d_in[18];

    char* ws = (char*)d_ws;
    size_t off = 0;
    auto take = [&](size_t bytes) -> char* {
        char* p = ws + off;
        off += (bytes + 255) & ~(size_t)255;
        return p;
    };
    _Float16* X16   = (_Float16*)take((size_t)MROWS * DMODEL * 2);
    _Float16* WqT   = (_Float16*)take((size_t)DMODEL * DMODEL * 2);
    _Float16* WkT   = (_Float16*)take((size_t)DMODEL * DMODEL * 2);
    _Float16* WvT   = (_Float16*)take((size_t)DMODEL * DMODEL * 2);
    _Float16* WoT   = (_Float16*)take((size_t)DMODEL * DMODEL * 2);
    _Float16* W1T   = (_Float16*)take((size_t)DMODEL * DFF * 2);
    _Float16* W2T   = (_Float16*)take((size_t)DFF * DMODEL * 2);
    _Float16* Qh    = (_Float16*)take((size_t)MROWS * DMODEL * 2);
    _Float16* Khd   = (_Float16*)take((size_t)MROWS * DMODEL * 2);
    _Float16* VtT   = (_Float16*)take((size_t)MROWS * DMODEL * 2);
    _Float16* AttnX = (_Float16*)take((size_t)MROWS * DMODEL * 2);
    float*    Y1    = (float*)   take((size_t)MROWS * DMODEL * 4);
    float*    H1f32 = (float*)   take((size_t)MROWS * DMODEL * 4);
    _Float16* H1f16 = (_Float16*)take((size_t)MROWS * DMODEL * 2);
    _Float16* FF1   = (_Float16*)take((size_t)MROWS * DFF * 2);
    float*    Y2    = (float*)   take((size_t)MROWS * DMODEL * 4);

    // 1) casts / weight transposes
    cvt_f16_kernel<<<(MROWS * DMODEL + 255) / 256, 256, 0, stream>>>(src, X16, MROWS * DMODEL);
    cvt_transpose_kernel<<<(DMODEL * DMODEL + 255) / 256, 256, 0, stream>>>(Wq, WqT, DMODEL, DMODEL);
    cvt_transpose_kernel<<<(DMODEL * DMODEL + 255) / 256, 256, 0, stream>>>(Wk, WkT, DMODEL, DMODEL);
    cvt_transpose_kernel<<<(DMODEL * DMODEL + 255) / 256, 256, 0, stream>>>(Wv, WvT, DMODEL, DMODEL);
    cvt_transpose_kernel<<<(DMODEL * DMODEL + 255) / 256, 256, 0, stream>>>(Wo, WoT, DMODEL, DMODEL);
    cvt_transpose_kernel<<<(DMODEL * DFF + 255) / 256, 256, 0, stream>>>(W1, W1T, DMODEL, DFF);
    cvt_transpose_kernel<<<(DFF * DMODEL + 255) / 256, 256, 0, stream>>>(W2, W2T, DFF, DMODEL);

    // 2) QKV projections (split heads; V transposed for PV B-fragments)
    dim3 gQKV(DMODEL / 128, MROWS / 128);
    gemm_f16_kernel<<<gQKV, 256, 0, stream>>>(X16, WqT, bq, nullptr, Qh,  MROWS, DMODEL, DMODEL, 3);
    gemm_f16_kernel<<<gQKV, 256, 0, stream>>>(X16, WkT, bk, nullptr, Khd, MROWS, DMODEL, DMODEL, 3);
    gemm_f16_kernel<<<gQKV, 256, 0, stream>>>(X16, WvT, bv, nullptr, VtT, MROWS, DMODEL, DMODEL, 4);

    // 3) flash attention
    dim3 gA(SEQ / 128, 2 * NHEAD);
    attn_kernel<<<gA, 256, 0, stream>>>(Qh, Khd, VtT, pad, AttnX);

    // 4) O-projection + residual, then LN1 (f32 + f16 copies)
    gemm_f16_kernel<<<gQKV, 256, 0, stream>>>(AttnX, WoT, bo, src, Y1, MROWS, DMODEL, DMODEL, 2);
    ln_kernel<<<MROWS, 256, 0, stream>>>(Y1, gamma1, beta1, H1f32, H1f16, nullptr);

    // 5) FFN: GEMM+GELU, GEMM+residual, final LN with padding zero
    dim3 gF1(DFF / 128, MROWS / 128);
    gemm_f16_kernel<<<gF1, 256, 0, stream>>>(H1f16, W1T, b1, nullptr, FF1, MROWS, DFF, DMODEL, 1);
    gemm_f16_kernel<<<gQKV, 256, 0, stream>>>(FF1, W2T, b2, H1f32, Y2, MROWS, DMODEL, DFF, 2);
    ln_kernel<<<MROWS, 256, 0, stream>>>(Y2, gamma2, beta2, (float*)d_out, nullptr, pad);
}
